// MyLSTM_68882685493761
// MI455X (gfx1250) — matrix operand order repaired and checked by reading, not compiled, not run on hardware
//
#include <hip/hip_runtime.h>

// ---------------------------------------------------------------------------
// Types for CDNA5 WMMA (wave32): 16x16x32 f16 -> f32 accumulate
// ---------------------------------------------------------------------------
typedef __attribute__((ext_vector_type(16))) _Float16 v16h;
typedef __attribute__((ext_vector_type(8)))  _Float16 h8;
typedef __attribute__((ext_vector_type(4)))  _Float16 h4;
typedef __attribute__((ext_vector_type(8)))  float    v8f;
typedef __attribute__((ext_vector_type(4)))  float    v4f;
typedef __attribute__((ext_vector_type(4)))  int      v4i;

#define AS1 __attribute__((address_space(1)))
#define AS3 __attribute__((address_space(3)))

#define T_STEPS 16
#define BATCH   128

// ---------------------------------------------------------------------------
// Async global->LDS copy of one 16B chunk (per lane), with graceful fallback.
// Builtin signature (from compiler diagnostic): (AS1 int4*, AS3 int4*, Ii, Ii)
// ---------------------------------------------------------------------------
__device__ __forceinline__ void async_copy16(const _Float16* g, _Float16* l) {
#if __has_builtin(__builtin_amdgcn_global_load_async_to_lds_b128)
  __builtin_amdgcn_global_load_async_to_lds_b128((AS1 v4i*)g, (AS3 v4i*)l, 0, 0);
#else
  *reinterpret_cast<h8*>(l) = *reinterpret_cast<const h8*>(g);
#endif
}

__device__ __forceinline__ void wait_async() {
#if __has_builtin(__builtin_amdgcn_global_load_async_to_lds_b128)
#if __has_builtin(__builtin_amdgcn_s_wait_asynccnt)
  __builtin_amdgcn_s_wait_asynccnt(0);
#else
  asm volatile("s_wait_asynccnt 0" ::: "memory");
#endif
#endif
}

// ---------------------------------------------------------------------------
// Fragment loaders; A and B share a contiguous k order (k = 16*(lane/16) + element).
// A (16x32, M x K) from row-major [*, ld]:
//   lanes 0-15  (M=lane):    halves = K[0..15]
//   lanes 16-31 (M=lane-16): halves = K[16..31]
// B (32x16, K x N) taken as W[N,K] row-major (B = W^T):
//   lanes 0-15  (N=lane):    halves = K[0..15]
//   lanes 16-31 (N=lane-16): halves = K[16..31]
// ---------------------------------------------------------------------------
__device__ __forceinline__ v16h load_frag_a(const _Float16* __restrict__ p,
                                            int ld, int row0, int k0, int lane) {
  const int r  = row0 + (lane & 15);
  const int kk = k0 + ((lane & 16) ? 16 : 0);
  const _Float16* base = p + (size_t)r * ld + kk;
  union { v16h v; h8 h[2]; } u;
  u.h[0] = *reinterpret_cast<const h8*>(base);        // K +0..7
  u.h[1] = *reinterpret_cast<const h8*>(base + 8);    // K +8..15
  return u.v;
}

__device__ __forceinline__ v16h load_frag_b(const _Float16* __restrict__ w,
                                            int ld, int n0, int k0, int lane) {
  const int n  = n0 + (lane & 15);
  const int kk = k0 + ((lane & 16) ? 16 : 0);
  const _Float16* base = w + (size_t)n * ld + kk;
  union { v16h v; h8 h[2]; } u;
  u.h[0] = *reinterpret_cast<const h8*>(base);        // K +0..7
  u.h[1] = *reinterpret_cast<const h8*>(base + 8);    // K +8..15
  return u.v;
}

// B fragment from an LDS panel: panel[row][0..GK) halves, row stride BSTRIDE
#define GK      64                 // K halves per staged panel
#define BPAD    8                  // pad halves per row (16B) -> bank spread
#define BSTRIDE (GK + BPAD)        // 72 halves = 144B per row

__device__ __forceinline__ v16h load_frag_b_lds(const _Float16* panel,
                                                int n0, int k0, int lane) {
  const int n  = n0 + (lane & 15);
  const int kk = k0 + ((lane & 16) ? 16 : 0);
  const _Float16* base = panel + n * BSTRIDE + kk;
  union { v16h v; h8 h[2]; } u;
  u.h[0] = *reinterpret_cast<const h8*>(base);        // ds_load_b128
  u.h[1] = *reinterpret_cast<const h8*>(base + 8);
  return u.v;
}

// ---------------------------------------------------------------------------
// f32 -> f16 conversion (vectorized x4)
// ---------------------------------------------------------------------------
__global__ void cvt_f32_to_f16(const float* __restrict__ in,
                               _Float16* __restrict__ out, int n4) {
  int i = blockIdx.x * blockDim.x + threadIdx.x;
  if (i >= n4) return;
  v4f v = reinterpret_cast<const v4f*>(in)[i];
  h4 o;
  o[0] = (_Float16)v[0]; o[1] = (_Float16)v[1];
  o[2] = (_Float16)v[2]; o[3] = (_Float16)v[3];
  reinterpret_cast<h4*>(out)[i] = o;
}

// ---------------------------------------------------------------------------
// GEMM:  C[M,N] (f32) = A[M,K] (f16) * W[N,K]^T (f16) + bias[N]
// Block tile: 128(M) x 64(N); wave w -> rows [bm*128 + 16w, +16), 4 n-subtiles.
// Shared 64-row B panel is double-buffered in LDS via async-to-LDS copies
// (ASYNCcnt), fragments read back with ds_load_b128. A fragments stream from
// global (L2-resident, reused across n-groups).
// ---------------------------------------------------------------------------
__global__ void __launch_bounds__(256)
gemm_f16f32_bias(const _Float16* __restrict__ A, const _Float16* __restrict__ W,
                 const float* __restrict__ bias, float* __restrict__ C,
                 int M, int N, int K) {
  __shared__ _Float16 bpanel[2][64 * BSTRIDE];   // 2 x 9216B

  const int tid  = threadIdx.x;
  const int lane = tid & 31;
  const int wv   = tid >> 5;                     // 0..7
  const int nGroups = N >> 6;
  const int bm = (int)blockIdx.x / nGroups;
  const int ng = (int)blockIdx.x % nGroups;
  const int m0 = bm * 128 + wv * 16;
  const int n0 = ng * 64;

  // Stage one 64-row x GK-half panel: 512 16B chunks, 2 per thread.
  auto stage = [&](int buf, int k0) {
#pragma unroll
    for (int c = 0; c < 2; ++c) {
      const int chunk = tid + c * 256;           // 0..511
      const int row   = chunk >> 3;              // 8 chunks per row
      const int co    = (chunk & 7) << 3;        // half offset in row
      const _Float16* g = W + (size_t)(n0 + row) * K + (k0 + co);
      _Float16* l = &bpanel[buf][row * BSTRIDE + co];
      async_copy16(g, l);
    }
  };

  v8f z = {};
  v8f acc[4] = {z, z, z, z};

  const int nk = K / GK;
  stage(0, 0);
  for (int kt = 0; kt < nk; ++kt) {
    wait_async();                 // this wave's panel chunks have landed
    __syncthreads();              // => whole panel visible to all waves
    if (kt + 1 < nk) stage((kt + 1) & 1, (kt + 1) * GK);
    const _Float16* panel = bpanel[kt & 1];
#pragma unroll
    for (int kh = 0; kh < GK; kh += 32) {
      v16h a = load_frag_a(A, K, m0, kt * GK + kh, lane);
#pragma unroll
      for (int s = 0; s < 4; ++s) {
        v16h b = load_frag_b_lds(panel, s << 4, kh, lane);
        acc[s] = __builtin_amdgcn_wmma_f32_16x16x32_f16(
            false, a, false, b, (short)0, acc[s], false, false);
      }
    }
    __syncthreads();              // done reading buf before it is re-staged
  }

  // C/D layout: lane L, VGPR r -> (M = m0 + r + 8*(L>=16), N = n0 + (L&15))
  const int rowOff = (lane & 16) ? 8 : 0;
#pragma unroll
  for (int s = 0; s < 4; ++s) {
    const int col = n0 + (s << 4) + (lane & 15);
    const float bv = bias ? bias[col] : 0.0f;
#pragma unroll
    for (int r = 0; r < 8; ++r) {
      C[(size_t)(m0 + r + rowOff) * N + col] = acc[s][r] + bv;
    }
  }
}

// ---------------------------------------------------------------------------
// Persistent fused LSTM recurrence.
//   pre  : [T, B, 4H] f32   (x-projection + bias, precomputed)
//   Whh  : [4H, H] f16      (PyTorch gate order i,f,g,o along rows)
//   hbuf : [2, B, H] f16    (ping-pong hidden state)
//   yout : [T, B, H]        (f16 if OUT_F16 else f32)
// Each wave owns a 16-batch x 16-h-column tile plus its 4 gate tiles
// (cols j, H+j, 2H+j, 3H+j): i/f/g/o land in this wave's accumulators and
// cell state c stays in registers across all 16 steps. Device-wide spin
// barrier between steps (all workgroups resident: <=32 blocks).
// ---------------------------------------------------------------------------
template <bool OUT_F16>
__global__ void __launch_bounds__(256)
lstm_recur(const float* __restrict__ pre, const _Float16* __restrict__ Whh,
           _Float16* __restrict__ hbuf, void* __restrict__ yout,
           unsigned* __restrict__ bar, int H, int nBlocks) {
  const int lane = threadIdx.x & 31;
  const int wave = (int)(blockIdx.x * blockDim.x + threadIdx.x) >> 5;
  const int Nj = H >> 4;            // # of 16-wide h-column tiles
  const int mt = wave / Nj;         // batch tile 0..7
  const int jt = wave % Nj;
  const int m0 = mt << 4;
  const int j0 = jt << 4;
  const int colJ   = j0 + (lane & 15);
  const int rowOff = (lane & 16) ? 8 : 0;
  const int G4 = 4 * H;

  float c[8];
#pragma unroll
  for (int r = 0; r < 8; ++r) c[r] = 0.0f;

  for (int t = 0; t < T_STEPS; ++t) {
    v8f z = {};
    v8f acc[4] = {z, z, z, z};
    const _Float16* hin  = hbuf + (size_t)((t & 1) ^ 1) * BATCH * H;
    _Float16*       hout = hbuf + (size_t)(t & 1) * BATCH * H;

    if (t > 0) {                    // h(0) == 0: skip the recurrent GEMM
      for (int k = 0; k < H; k += 32) {
        v16h a = load_frag_a(hin, H, m0, k, lane);
#pragma unroll
        for (int s = 0; s < 4; ++s) {
          v16h b = load_frag_b(Whh, H, s * H + j0, k, lane);
          acc[s] = __builtin_amdgcn_wmma_f32_16x16x32_f16(
              false, a, false, b, (short)0, acc[s], false, false);
        }
      }
    }

    const float* pt = pre + (size_t)t * BATCH * G4;
#pragma unroll
    for (int r = 0; r < 8; ++r) {
      const int brow = m0 + r + rowOff;           // batch index
      const float* pr = pt + (size_t)brow * G4 + colJ;
      float gi = acc[0][r] + pr[0];
      float gf = acc[1][r] + pr[H];
      float gg = acc[2][r] + pr[2 * H];
      float go = acc[3][r] + pr[3 * H];
      float si = 1.0f / (1.0f + __expf(-gi));
      float sf = 1.0f / (1.0f + __expf(-gf));
      float so = 1.0f / (1.0f + __expf(-go));
      float c2 = sf * c[r] + si * tanhf(gg);
      float h2 = so * tanhf(c2);
      c[r] = c2;
      hout[(size_t)brow * H + colJ] = (_Float16)h2;
      if (OUT_F16)
        reinterpret_cast<_Float16*>(yout)[((size_t)t * BATCH + brow) * H + colJ] =
            (_Float16)h2;
      else
        reinterpret_cast<float*>(yout)[((size_t)t * BATCH + brow) * H + colJ] = h2;
    }

    if (t + 1 < T_STEPS) {          // device-wide barrier between steps
      __threadfence();
      __syncthreads();
      if (threadIdx.x == 0) {
        __hip_atomic_fetch_add(bar, 1u, __ATOMIC_RELEASE, __HIP_MEMORY_SCOPE_AGENT);
        const unsigned target = (unsigned)nBlocks * (unsigned)(t + 1);
        while (__hip_atomic_load(bar, __ATOMIC_ACQUIRE, __HIP_MEMORY_SCOPE_AGENT) <
               target) {
          __builtin_amdgcn_s_sleep(1);
        }
      }
      __syncthreads();
    }
  }
}

// ---------------------------------------------------------------------------
extern "C" void kernel_launch(void* const* d_in, const int* in_sizes, int n_in,
                              void* d_out, int out_size, void* d_ws, size_t ws_size,
                              hipStream_t stream) {
  (void)in_sizes; (void)n_in; (void)out_size; (void)ws_size;
  constexpr int T = 16, IN = 12288, H1 = 256, H2 = 512, B = 128;

  const float* inp  = (const float*)d_in[0];
  const float* Wih1 = (const float*)d_in[1];
  const float* Whh1 = (const float*)d_in[2];
  const float* b1   = (const float*)d_in[3];
  const float* Wih2 = (const float*)d_in[4];
  const float* Whh2 = (const float*)d_in[5];
  const float* b2   = (const float*)d_in[6];

  // Workspace carving (256B aligned)
  char* w = (char*)d_ws;
  auto carve = [&](size_t bytes) {
    char* p = w;
    w += (bytes + 255) & ~(size_t)255;
    return p;
  };
  _Float16* x16     = (_Float16*)carve((size_t)T * B * IN * 2);
  _Float16* Wih1_16 = (_Float16*)carve((size_t)4 * H1 * IN * 2);
  _Float16* Whh1_16 = (_Float16*)carve((size_t)4 * H1 * H1 * 2);
  _Float16* Wih2_16 = (_Float16*)carve((size_t)4 * H2 * H1 * 2);
  _Float16* Whh2_16 = (_Float16*)carve((size_t)4 * H2 * H2 * 2);
  float*    pre1    = (float*)carve((size_t)T * B * 4 * H1 * 4);
  float*    pre2    = (float*)carve((size_t)T * B * 4 * H2 * 4);
  _Float16* y1_16   = (_Float16*)carve((size_t)T * B * H1 * 2);
  _Float16* hb1     = (_Float16*)carve((size_t)2 * B * H1 * 2);
  _Float16* hb2     = (_Float16*)carve((size_t)2 * B * H2 * 2);
  unsigned* bars    = (unsigned*)carve(512);

  (void)hipMemsetAsync(bars, 0, 512, stream);  // barrier counters (capture-legal)

  auto cvt = [&](const float* src, _Float16* dst, size_t n) {
    int n4 = (int)(n >> 2);
    cvt_f32_to_f16<<<(n4 + 255) / 256, 256, 0, stream>>>(src, dst, n4);
  };
  cvt(inp,  x16,     (size_t)T * B * IN);      // flat reshape: already [T*B, IN]
  cvt(Wih1, Wih1_16, (size_t)4 * H1 * IN);
  cvt(Whh1, Whh1_16, (size_t)4 * H1 * H1);
  cvt(Wih2, Wih2_16, (size_t)4 * H2 * H1);
  cvt(Whh2, Whh2_16, (size_t)4 * H2 * H2);

  // Layer 1 input projection: [2048,12288] x [1024,12288]^T -> pre1
  {
    int M = T * B, N = 4 * H1, K = IN;
    int blocks = (M / 128) * (N / 64);         // 16 x 16 = 256 blocks
    gemm_f16f32_bias<<<blocks, 256, 0, stream>>>(x16, Wih1_16, b1, pre1, M, N, K);
  }
  // Layer 1 recurrence (persistent, 16 blocks resident)
  {
    int Nj = H1 / 16;  // 16 blocks * 8 waves == 8*Nj waves
    lstm_recur<true><<<Nj, 256, 0, stream>>>(pre1, Whh1_16, hb1, y1_16, bars, H1, Nj);
  }
  // Layer 2 input projection: [2048,256] x [2048,256]^T -> pre2
  {
    int M = T * B, N = 4 * H2, K = H1;
    int blocks = (M / 128) * (N / 64);         // 16 x 32 = 512 blocks
    gemm_f16f32_bias<<<blocks, 256, 0, stream>>>(y1_16, Wih2_16, b2, pre2, M, N, K);
  }
  // Layer 2 recurrence -> d_out (f32, flat [T,B,H2] order == final reshape)
  {
    int Nj = H2 / 16;  // 32 blocks
    lstm_recur<false><<<Nj, 256, 0, stream>>>(pre2, Whh2_16, hb2, d_out,
                                              bars + 64, H2, Nj);
  }
}
